// MultiHeadAttention_85598698209918
// MI455X (gfx1250) — matrix operand order repaired and checked by reading, not compiled, not run on hardware
//
#include <hip/hip_runtime.h>

#define D_MODEL 1024
#define NHEAD   16
#define HD      64
#define SEQ     2048
#define BATCH   2
#define NTOK    (BATCH * SEQ)   // 4096
#define QKVN    3072

typedef __bf16 bf16_t;
typedef bf16_t v16bf __attribute__((ext_vector_type(16)));
typedef float  v8f   __attribute__((ext_vector_type(8)));
typedef unsigned int v4u  __attribute__((ext_vector_type(4)));
typedef unsigned int u32x4 __attribute__((ext_vector_type(4)));
typedef int i32x8 __attribute__((ext_vector_type(8)));
typedef int i32x4 __attribute__((ext_vector_type(4)));

union Frag {
    v16bf bf;
    unsigned int u32[8];
    unsigned short u16[16];
    v4u q4[2];
};

__device__ __forceinline__ unsigned short f2bf(float f) {
    unsigned int u = __float_as_uint(f);
    u += 0x7FFFu + ((u >> 16) & 1u);      // round-to-nearest-even
    return (unsigned short)(u >> 16);
}

// Low 32 bits of a generic pointer to LDS = LDS byte offset (flat aperture rule).
__device__ __forceinline__ unsigned lds_off(const void* p) {
    return (unsigned)(size_t)p;
}

// ---------------------------------------------------------------------------
// TDM: issue a 2D bf16 tile load Global -> LDS via Tensor Data Mover.
// D# per CDNA5 ISA. tensor_dim == tile_dim (never OOB),
// LDS row padding via pad_interval/pad_amount (DWORD units).
// ---------------------------------------------------------------------------
__device__ __forceinline__ void tdm_load_2d(
    unsigned ldsOff, const unsigned short* gptr,
    unsigned tile_w, unsigned tile_h,     // elements (bf16): dim0 (contig), dim1 (rows)
    unsigned row_stride,                  // tensor_dim0_stride in elements
    unsigned pad_interval_code, unsigned pad_amount_code)
{
    const unsigned long long ga = (unsigned long long)(size_t)gptr;
    u32x4 g0;
    g0[0] = 1u;                                              // count=1 valid user D#
    g0[1] = ldsOff;                                          // lds_addr
    g0[2] = (unsigned)(ga & 0xFFFFFFFFu);                    // global_addr[31:0]
    g0[3] = (unsigned)((ga >> 32) & 0x01FFFFFFu) | (2u << 30); // addr[56:32] | type=2

    i32x8 g1;
    g1[0] = (int)((1u << 16)                                 // data_size = 2 bytes
                | (1u << 20)                                 // pad_enable
                | (pad_interval_code << 22)
                | (pad_amount_code << 25));
    g1[1] = (int)((tile_w & 0xFFFFu) << 16);                 // tensor_dim0[15:0]
    g1[2] = (int)(((tile_w >> 16) & 0xFFFFu) |
                  ((tile_h & 0xFFFFu) << 16));               // tensor_dim0 hi | dim1 lo
    g1[3] = (int)(((tile_h >> 16) & 0xFFFFu) |
                  (tile_w << 16));                           // dim1 hi | tile_dim0
    g1[4] = (int)(tile_h & 0xFFFFu);                         // tile_dim1 | tile_dim2=0
    g1[5] = (int)row_stride;                                 // tensor_dim0_stride[31:0]
    g1[6] = 0;                                               // stride0 hi | stride1 lo
    g1[7] = 0;

    const i32x4 z4 = {0, 0, 0, 0};
    const i32x8 z8 = {0, 0, 0, 0, 0, 0, 0, 0};
    // clang-23 / therock-10.0 6-arg form (2D: tile_dim2=0, extra groups zero)
    __builtin_amdgcn_tensor_load_to_lds(g0, g1, z4, z4, z8, 0);
}

// ---------------------------------------------------------------------------
// bf16 GEMM:  C[M,N] = A[M,K] * B[K,N] (+bias on f32 path)
// WG = 256 threads = 8 waves (4 x 2), tile 64 x 128, K-step 32.
// Tiles staged by TDM (wave 0 issues, double-buffered, TENSORcnt + barrier).
// ---------------------------------------------------------------------------
template <bool OUT_BF16>
__global__ __launch_bounds__(256) void gemm_bf16_wmma(
    const unsigned short* __restrict__ A,
    const unsigned short* __restrict__ B,
    void* __restrict__ Cp,
    const float* __restrict__ bias,
    int M, int N, int K)
{
    __shared__ unsigned short As[2][64][40];    // row stride 80B (16B aligned)
    __shared__ unsigned short Bs[2][32][136];   // row stride 272B

    const int tid  = threadIdx.x;
    const int wave = tid >> 5, lane = tid & 31;
    const int wm = wave & 3, wn = wave >> 2;
    const int half = lane >> 4, l16 = lane & 15;
    const int mBase = blockIdx.y * 64;
    const int nBase = blockIdx.x * 128;

    v8f acc[4];
#pragma unroll
    for (int c = 0; c < 4; ++c)
#pragma unroll
        for (int r = 0; r < 8; ++r) acc[c][r] = 0.f;

    auto issueAB = [&](int bufi, int kk0) {
        tdm_load_2d(lds_off(&As[bufi][0][0]),
                    A + (size_t)mBase * K + kk0, 32, 64, K, 3, 3);
        tdm_load_2d(lds_off(&Bs[bufi][0][0]),
                    B + (size_t)kk0 * N + nBase, 128, 32, N, 5, 3);
    };

    if (wave == 0) { issueAB(0, 0); __builtin_amdgcn_s_wait_tensorcnt(0); }
    __syncthreads();

    int buf = 0;
    for (int k0 = 0; k0 < K; k0 += 32) {
        if (wave == 0 && k0 + 32 < K) issueAB(buf ^ 1, k0 + 32);

        // A fragment: lane = M row, VGPR v packs K pair (v/4)*16 + half*8 + (v%4)*2
        Frag af;
        {
            const int m = wm * 16 + l16;
#pragma unroll
            for (int v = 0; v < 8; ++v) {
                const int kk = ((v >> 2) << 4) | (half << 3) | ((v & 3) << 1);
                af.u32[v] = *(const unsigned int*)&As[buf][m][kk];
            }
        }
#pragma unroll
        for (int c = 0; c < 4; ++c) {
            Frag bfr;   // B fragment: lane = N col, element s = K (s>>3)*16 + half*8 + (s&7)
            const int col = wn * 64 + c * 16;
#pragma unroll
            for (int s2 = 0; s2 < 16; ++s2)
                bfr.u16[s2] = Bs[buf][((s2 >> 3) << 4) | (half << 3) | (s2 & 7)][col + l16];
            acc[c] = __builtin_amdgcn_wmma_f32_16x16x32_bf16(
                false, af.bf, false, bfr.bf, (short)0, acc[c], false, false);
        }

        if (wave == 0) __builtin_amdgcn_s_wait_tensorcnt(0);
        __syncthreads();
        buf ^= 1;
    }

#pragma unroll
    for (int c = 0; c < 4; ++c) {
        const int col = nBase + wn * 64 + c * 16 + l16;
#pragma unroll
        for (int r = 0; r < 8; ++r) {
            const int row = mBase + wm * 16 + r + 8 * half;
            const float v = acc[c][r];
            if (OUT_BF16) {
                ((unsigned short*)Cp)[(size_t)row * N + col] = f2bf(v);
            } else {
                ((float*)Cp)[(size_t)row * N + col] = v + bias[col];
            }
        }
    }
}

// ---------------------------------------------------------------------------
// K transpose: qkv K-block [token][1024] -> ktg [b*1024 + hcol][seq] (bf16)
// ---------------------------------------------------------------------------
__global__ __launch_bounds__(256) void transpose_k(
    const unsigned short* __restrict__ qkv, unsigned short* __restrict__ ktg)
{
    __shared__ unsigned short t[32][34];
    const int b = blockIdx.z;
    const int x0 = blockIdx.x * 32;     // hcol
    const int y0 = blockIdx.y * 32;     // seq
#pragma unroll
    for (int i = 0; i < 4; ++i) {
        const int y = threadIdx.y + i * 8;
        t[y][threadIdx.x] =
            qkv[(size_t)(b * SEQ + y0 + y) * QKVN + D_MODEL + x0 + threadIdx.x];
    }
    __syncthreads();
#pragma unroll
    for (int i = 0; i < 4; ++i) {
        const int yy = threadIdx.y + i * 8;
        ktg[(size_t)(b * D_MODEL + x0 + yy) * SEQ + y0 + threadIdx.x] =
            t[threadIdx.x][yy];
    }
}

// ---------------------------------------------------------------------------
// Flash attention: WG = 128 threads = 4 waves, 16 queries/wave of one (b,h).
// K^T and V tiles staged by TDM, double-buffered. Scores are pre-scaled
// (1/sqrt(D_OUT) folded into Wq), so no per-step scaling here.
// ---------------------------------------------------------------------------
__global__ __launch_bounds__(128) void attn_wmma(
    const unsigned short* __restrict__ qkv,   // [token][3072]
    const unsigned short* __restrict__ ktg,   // [b*1024 + hcol][seq]
    unsigned short* __restrict__ ctx)         // [token][1024]
{
    __shared__ unsigned short KtT[2][64][40];   // [hd][key], row stride 80B
    __shared__ unsigned short Vt[2][32][72];    // [key][hd], row stride 144B
    __shared__ unsigned short Ps[4][16][32];    // per-wave P relayout scratch

    const int tid  = threadIdx.x;
    const int wave = tid >> 5, lane = tid & 31;
    const int half = lane >> 4, l16 = lane & 15;
    const int qblock = blockIdx.x;          // 0..31 (64 queries per WG)
    const int bh = blockIdx.y;
    const int b = bh >> 4, h = bh & 15;
    const int qseq = qblock * 64 + wave * 16;
    const size_t tokBase = (size_t)b * SEQ;

    // Q fragments (hd 0-31, 32-63), loaded once
    Frag qf[2];
    {
        const unsigned short* qrow = qkv + (tokBase + qseq + l16) * QKVN + h * HD;
#pragma unroll
        for (int cc = 0; cc < 2; ++cc)
#pragma unroll
            for (int v = 0; v < 8; ++v) {
                const int kk = cc * 32 + (((v >> 2) << 4) | (half << 3) | ((v & 3) << 1));
                qf[cc].u32[v] = *(const unsigned int*)(qrow + kk);
            }
    }

    float mi[8], li[8];
    v8f O[4];
#pragma unroll
    for (int r = 0; r < 8; ++r) { mi[r] = -3.0e38f; li[r] = 0.f; }
#pragma unroll
    for (int c = 0; c < 4; ++c)
#pragma unroll
        for (int r = 0; r < 8; ++r) O[c][r] = 0.f;

    auto issueKV = [&](int bufi, int kb2) {
        tdm_load_2d(lds_off(&KtT[bufi][0][0]),
                    ktg + (size_t)(b * D_MODEL + h * HD) * SEQ + kb2,
                    32, 64, SEQ, 3, 3);
        tdm_load_2d(lds_off(&Vt[bufi][0][0]),
                    qkv + (tokBase + kb2) * QKVN + 2 * D_MODEL + h * HD,
                    64, 32, QKVN, 4, 3);
    };

    const int nsteps = (qblock + 1) * 2;    // keys 0 .. (qblock+1)*64-1
    if (wave == 0) { issueKV(0, 0); __builtin_amdgcn_s_wait_tensorcnt(0); }
    __syncthreads();

    int buf = 0;
    for (int s = 0; s < nsteps; ++s) {
        const int kb = s * 32;
        if (wave == 0 && s + 1 < nsteps) issueKV(buf ^ 1, kb + 32);

        // scores S[j] (16q x 16k), contract hd in 2 chunks of 32 (pre-scaled)
        v8f S[2];
#pragma unroll
        for (int j = 0; j < 2; ++j) {
#pragma unroll
            for (int r = 0; r < 8; ++r) S[j][r] = 0.f;
#pragma unroll
            for (int cc = 0; cc < 2; ++cc) {
                Frag kf;   // B = K^T: lane = key col, element s = hd (s>>3)*16 + half*8 + (s&7)
#pragma unroll
                for (int s2 = 0; s2 < 16; ++s2)
                    kf.u16[s2] = KtT[buf][cc * 32 + (((s2 >> 3) << 4) | (half << 3) | (s2 & 7))][j * 16 + l16];
                S[j] = __builtin_amdgcn_wmma_f32_16x16x32_bf16(
                    false, qf[cc].bf, false, kf.bf, (short)0, S[j], false, false);
            }
        }

        // causal mask only on boundary steps (wave-uniform predicate)
        if (kb + 31 > qseq) {
#pragma unroll
            for (int r = 0; r < 8; ++r) {
                const int qg = qseq + r + 8 * half;
#pragma unroll
                for (int j = 0; j < 2; ++j) {
                    const int kg = kb + j * 16 + l16;
                    if (kg > qg) S[j][r] = -3.0e38f;
                }
            }
        }

        // online softmax
        float pm[8];
#pragma unroll
        for (int r = 0; r < 8; ++r) pm[r] = fmaxf(S[0][r], S[1][r]);
#pragma unroll
        for (int off = 8; off >= 1; off >>= 1)
#pragma unroll
            for (int r = 0; r < 8; ++r)
                pm[r] = fmaxf(pm[r], __shfl_xor(pm[r], off, 16));

        float rs[8], alpha[8];
#pragma unroll
        for (int r = 0; r < 8; ++r) {
            const float nm = fmaxf(mi[r], pm[r]);
            alpha[r] = __expf(mi[r] - nm);
            const float p0 = __expf(S[0][r] - nm);
            const float p1 = __expf(S[1][r] - nm);
            S[0][r] = p0; S[1][r] = p1;
            rs[r] = p0 + p1;
            mi[r] = nm;
        }
#pragma unroll
        for (int off = 8; off >= 1; off >>= 1)
#pragma unroll
            for (int r = 0; r < 8; ++r)
                rs[r] += __shfl_xor(rs[r], off, 16);
#pragma unroll
        for (int r = 0; r < 8; ++r) li[r] = li[r] * alpha[r] + rs[r];
#pragma unroll
        for (int c = 0; c < 4; ++c)
#pragma unroll
            for (int r = 0; r < 8; ++r) O[c][r] *= alpha[r];

        // P: C layout -> A layout via per-wave LDS scratch (in-order per wave)
#pragma unroll
        for (int j = 0; j < 2; ++j)
#pragma unroll
            for (int r = 0; r < 8; ++r)
                Ps[wave][r + 8 * half][j * 16 + l16] = f2bf(S[j][r]);
        asm volatile("" ::: "memory");

        Frag pf;
#pragma unroll
        for (int v = 0; v < 8; ++v) {
            const int kk = ((v >> 2) << 4) | (half << 3) | ((v & 3) << 1);
            pf.u32[v] = *(const unsigned int*)&Ps[wave][l16][kk];
        }

        // O += P(16x32) * V(32x64)
#pragma unroll
        for (int c = 0; c < 4; ++c) {
            Frag vf;
#pragma unroll
            for (int s2 = 0; s2 < 16; ++s2)
                vf.u16[s2] = Vt[buf][((s2 >> 3) << 4) | (half << 3) | (s2 & 7)][c * 16 + l16];
            O[c] = __builtin_amdgcn_wmma_f32_16x16x32_bf16(
                false, pf.bf, false, vf.bf, (short)0, O[c], false, false);
        }

        if (wave == 0) __builtin_amdgcn_s_wait_tensorcnt(0);
        __syncthreads();
        buf ^= 1;
    }

#pragma unroll
    for (int r = 0; r < 8; ++r) li[r] = 1.0f / li[r];
#pragma unroll
    for (int c = 0; c < 4; ++c) {
        const int col = h * HD + c * 16 + l16;
#pragma unroll
        for (int r = 0; r < 8; ++r) {
            const size_t row = tokBase + qseq + r + 8 * half;
            ctx[row * D_MODEL + col] = f2bf(O[c][r] * li[r]);
        }
    }
}

// ---------------------------------------------------------------------------
// f32 -> bf16 conversion kernels
// ---------------------------------------------------------------------------
__global__ __launch_bounds__(256) void cvt_f32_bf16(
    const float* __restrict__ in, unsigned short* __restrict__ out, int n)
{
    const int i = blockIdx.x * 256 + threadIdx.x;
    if (i < n) out[i] = f2bf(in[i]);
}

__global__ __launch_bounds__(256) void cvt_qkvw(
    const float* __restrict__ Wq, const float* __restrict__ Wk,
    const float* __restrict__ Wv, unsigned short* __restrict__ out)
{
    const int i = blockIdx.x * 256 + threadIdx.x;   // over 1024*3072
    const int row = i / QKVN, col = i - row * QKVN;
    const int p = col >> 10, j = col & 1023;
    const float* W = (p == 0) ? Wq : ((p == 1) ? Wk : Wv);
    float v = W[row * D_MODEL + j];
    if (p == 0) v *= 0.03125f;      // fold 1/sqrt(D_OUT) into Wq (exact: 2^-5)
    out[i] = f2bf(v);
}

// ---------------------------------------------------------------------------
extern "C" void kernel_launch(void* const* d_in, const int* in_sizes, int n_in,
                              void* d_out, int out_size, void* d_ws, size_t ws_size,
                              hipStream_t stream)
{
    (void)in_sizes; (void)n_in; (void)out_size; (void)ws_size;
    const float* x  = (const float*)d_in[0];
    const float* Wq = (const float*)d_in[1];
    const float* Wk = (const float*)d_in[2];
    const float* Wv = (const float*)d_in[3];
    const float* Wo = (const float*)d_in[4];
    const float* bo = (const float*)d_in[5];
    float* out = (float*)d_out;

    char* ws = (char*)d_ws;
    unsigned short* xb   = (unsigned short*)(ws);                       // 8 MB
    unsigned short* wqkv = (unsigned short*)(ws + 8388608);             // 6 MB
    unsigned short* wob  = (unsigned short*)(ws + 8388608 + 6291456);   // 2 MB
    unsigned short* qkvb = (unsigned short*)(ws + 16777216);            // 24 MB
    unsigned short* ctxb = (unsigned short*)(ws + 41943040);            // 8 MB
    unsigned short* ktg  = (unsigned short*)(ws + 50331648);            // 8 MB

    cvt_f32_bf16<<<(NTOK * D_MODEL) / 256, 256, 0, stream>>>(x, xb, NTOK * D_MODEL);
    cvt_qkvw<<<(D_MODEL * QKVN) / 256, 256, 0, stream>>>(Wq, Wk, Wv, wqkv);
    cvt_f32_bf16<<<(D_MODEL * D_MODEL) / 256, 256, 0, stream>>>(Wo, wob, D_MODEL * D_MODEL);

    // QKV projection: [4096 x 3072] = xb * wqkv
    gemm_bf16_wmma<true><<<dim3(QKVN / 128, NTOK / 64), 256, 0, stream>>>(
        xb, wqkv, qkvb, nullptr, NTOK, QKVN, D_MODEL);

    // K -> K^T (per batch): [seq][1024] -> [1024][seq]
    transpose_k<<<dim3(D_MODEL / 32, SEQ / 32, BATCH), dim3(32, 8), 0, stream>>>(
        qkvb, ktg);

    // causal flash attention per (b,h)
    attn_wmma<<<dim3(SEQ / 64, BATCH * NHEAD), 128, 0, stream>>>(qkvb, ktg, ctxb);

    // output projection + bias: f32 out
    gemm_bf16_wmma<false><<<dim3(D_MODEL / 128, NTOK / 64), 256, 0, stream>>>(
        ctxb, wob, out, bo, NTOK, D_MODEL, D_MODEL);
}
